// RWKV6Attention_85117661872245
// MI455X (gfx1250) — compile-verified, matrix operand order checked by reading, not run on hardware
//
#include <hip/hip_runtime.h>
#include <hip/hip_bf16.h>

// ---------------------------------------------------------------------------
// RWKV6 attention forward for MI455X (gfx1250, wave32).
//   B=2, T=2048, D=1024, H=4, Dk=128, Dv=256, KD=512, VD=1024, R5=160, LR=64
// GEMMs: v_wmma_f32_16x16x32_bf16, wave tile 64x64 (16 accumulators),
// 4 waves/block -> 256x64. Weight panel staged into LDS by the Tensor Data
// Mover, double-buffered (DMA of chunk c+1 overlaps WMMA on chunk c), with
// TDM row padding (128B + 16B -> 144B LDS stride) for bank-conflict-free
// ds_load_b128 fragment reads.
// Scan: k-dimension split 4 ways across blocks (state slice in VGPRs, decay
// precomputed in the w-GEMM epilogue), partials reduced deterministically in
// the post kernel.
// ---------------------------------------------------------------------------

#define B_  2
#define T_  2048
#define D_  1024
#define H_  4
#define DK_ 128
#define DV_ 256
#define KD_ 512
#define VD_ 1024
#define LR_ 64
#define R5_ 160
#define BT_ (B_*T_)

#define KC_  64    // K-chunk staged per TDM transfer
#define SBS_ 72    // padded LDS row stride in elements (64 + 8)
#define SPLIT_ 4   // k-dimension split of the scan
#define KSL_ (DK_/SPLIT_)  // 32 k-rows per scan block

typedef __attribute__((ext_vector_type(16))) __bf16 v16bf;
typedef __attribute__((ext_vector_type(8)))  __bf16 v8bf;
typedef __attribute__((ext_vector_type(8)))  float  v8f;
typedef __attribute__((ext_vector_type(4)))  unsigned int v4u;
typedef __attribute__((ext_vector_type(8)))  int v8i;
typedef __attribute__((ext_vector_type(4)))  int v4i;

// ---------------------------------------------------------------------------
// fp32 -> bf16 conversion
// ---------------------------------------------------------------------------
__global__ void cvt_f32_bf16(const float* __restrict__ src,
                             __bf16* __restrict__ dst, int n) {
  int i = blockIdx.x * blockDim.x + threadIdx.x;
  if (i < n) dst[i] = (__bf16)src[i];
}

// ---------------------------------------------------------------------------
// Pre-kernel: token shift, LerpLinear x_proj (D->160) + tanh, grouped einsum
// with xp_W2, emit the five bf16 mixed activation rows. One block per token.
// ---------------------------------------------------------------------------
__global__ __launch_bounds__(256) void pre_kernel(
    const float* __restrict__ x, const float* __restrict__ xp_mu,
    const float* __restrict__ xp_W1, const float* __restrict__ xp_W2,
    const float* __restrict__ x_bias,
    __bf16* __restrict__ Ar, __bf16* __restrict__ Aw, __bf16* __restrict__ Ak,
    __bf16* __restrict__ Av, __bf16* __restrict__ Ag) {
  __shared__ float sx[D_];
  __shared__ float sd[D_];
  __shared__ float sm[D_];
  __shared__ float sxl[R5_];
  const int blk = blockIdx.x;
  const int t = blk % T_;
  const float* xrow = x + (size_t)blk * D_;
  for (int d = threadIdx.x; d < D_; d += 256) {
    float xv = xrow[d];
    float sh = (t > 0) ? xrow[d - D_] : 0.f;  // token shift (zero pad)
    float dl = sh - xv;
    sx[d] = xv; sd[d] = dl;
    sm[d] = xv + dl * xp_mu[d];
  }
  __syncthreads();
  if (threadIdx.x < R5_) {
    const float* wrow = xp_W1 + (size_t)threadIdx.x * D_;
    float acc = 0.f;
    for (int d = 0; d < D_; ++d) acc += sm[d] * wrow[d];
    sxl[threadIdx.x] = tanhf(acc);
  }
  __syncthreads();
  __bf16* outs[5] = {Ar, Aw, Ak, Av, Ag};
  for (int d = threadIdx.x; d < D_; d += 256) {
    const float* w2row = xp_W2 + (size_t)d * R5_;
    float xv = sx[d], dl = sd[d];
#pragma unroll
    for (int n = 0; n < 5; ++n) {
      float mu = x_bias[n * D_ + d];
#pragma unroll
      for (int r = 0; r < 32; ++r) mu += sxl[n * 32 + r] * w2row[n * 32 + r];
      outs[n][(size_t)blk * D_ + d] = (__bf16)(xv + dl * mu);
    }
  }
}

// ---------------------------------------------------------------------------
// 16-bit A/B fragment loader per ISA 16x32 layout: lane (g=l>>4, idx=l&15)
// holds K in {8g..8g+7} U {16+8g..23+8g} -> two 16-byte loads.
// ---------------------------------------------------------------------------
__device__ inline v16bf load_frag(const __bf16* __restrict__ rowbase, int g) {
  v8bf lo = *reinterpret_cast<const v8bf*>(rowbase + g * 8);
  v8bf hi = *reinterpret_cast<const v8bf*>(rowbase + 16 + g * 8);
  return __builtin_shufflevector(lo, hi, 0, 1, 2, 3, 4, 5, 6, 7,
                                 8, 9, 10, 11, 12, 13, 14, 15);
}

// ---------------------------------------------------------------------------
// TDM descriptor (CDNA5 ISA Sec. 8): group1 is loop
// invariant (dims/strides/padding); group0 carries LDS + global addresses.
// data_size=2B, tile KC_ x 64, pad 16B per 128B row -> 144B LDS row stride.
// ---------------------------------------------------------------------------
__device__ inline v8i tdm_make_g1(unsigned ld_elems, unsigned rows) {
  unsigned w0 = (1u << 16) | (1u << 20) | (4u << 22) | (3u << 25);
  unsigned td0 = 0x40000000u, td1 = 0x40000000u;  // huge dims: no OOB clip
  unsigned tile0 = KC_, tile1 = rows;
  unsigned long long s0 = (unsigned long long)ld_elems;  // dim0 stride
  v8i g1;
  g1[0] = (int)w0;
  g1[1] = (int)((td0 & 0xffffu) << 16);
  g1[2] = (int)((td0 >> 16) | ((td1 & 0xffffu) << 16));
  g1[3] = (int)((td1 >> 16) | (tile0 << 16));
  g1[4] = (int)tile1;
  g1[5] = (int)(s0 & 0xffffffffu);
  g1[6] = (int)((s0 >> 32) & 0xffffu);
  g1[7] = 0;
  return g1;
}

__device__ inline void tdm_issue(const __bf16* gptr, unsigned lds_off,
                                 v8i g1) {
  unsigned long long ga = (unsigned long long)(uintptr_t)gptr;
  v4u g0;
  g0.x = 1u;                                     // count=1 (valid user D#)
  g0.y = lds_off;                                // LDS byte address
  g0.z = (unsigned)(ga & 0xffffffffu);           // global_addr[31:0]
  g0.w = (unsigned)((ga >> 32) & 0x1ffffffu)     // global_addr[56:32]
       | (2u << 30);                             // type=2 ("image")
  v4i z4 = {};
#if defined(__clang_major__) && (__clang_major__ >= 23)
  v8i z8 = {};
  __builtin_amdgcn_tensor_load_to_lds(g0, g1, z4, z4, z8, 0);
#else
  __builtin_amdgcn_tensor_load_to_lds(g0, g1, z4, z4, 0);
#endif
}

// ---------------------------------------------------------------------------
// WMMA bf16 GEMM:  C[M,N] = A[M,K] @ W[N,K]^T
// Block = 128 threads (4 waves); wave tile 64(M) x 64(N), 16 accumulators.
// grid = (N/64, M/256). Weight panel double-buffered through the TDM.
// Epilogue act: 0=none, 1=tanh, 2=exp(-exp(x))  (RWKV decay).
// ---------------------------------------------------------------------------
__global__ __launch_bounds__(128) void gemm_bf16_wmma(
    const __bf16* __restrict__ A, int lda,
    const __bf16* __restrict__ W, int ldw,
    float* __restrict__ Cf, __bf16* __restrict__ Cbf,
    const float* __restrict__ bias,
    int M, int N, int K, int act) {
  __shared__ __align__(16) __bf16 sB[2][64 * SBS_];  // 2 x 9216 B
  const int lane = threadIdx.x & 31;
  const int wave = threadIdx.x >> 5;
  const int g = lane >> 4, idx = lane & 15;
  const int m0 = blockIdx.y * 256 + wave * 64;
  const int n0 = blockIdx.x * 64;

  v8f acc[4][4];
#pragma unroll
  for (int mi = 0; mi < 4; ++mi)
#pragma unroll
    for (int j = 0; j < 4; ++j) acc[mi][j] = (v8f){};

  const __bf16* arow[4];
#pragma unroll
  for (int mi = 0; mi < 4; ++mi)
    arow[mi] = A + (size_t)(m0 + mi * 16 + idx) * lda;

  const v8i g1 = tdm_make_g1((unsigned)ldw, 64);
  const unsigned sB0 = (unsigned)(uintptr_t)&sB[0][0];
  const unsigned sB1 = (unsigned)(uintptr_t)&sB[1][0];
  const __bf16* wpanel = W + (size_t)n0 * ldw;

  if (threadIdx.x == 0) tdm_issue(wpanel, sB0, g1);  // chunk 0

  const int nchunk = K / KC_;
  for (int c = 0; c < nchunk; ++c) {
    const int kc = c * KC_;
    if (threadIdx.x == 0) {
      __builtin_amdgcn_s_wait_tensorcnt(0);           // chunk c landed
      if (c + 1 < nchunk)                             // overlap DMA of c+1
        tdm_issue(wpanel + kc + KC_, (c & 1) ? sB0 : sB1, g1);
    }
    __syncthreads();
    const __bf16* bbuf = &sB[c & 1][0];
    if (c + 1 < nchunk) {
#pragma unroll
      for (int mi = 0; mi < 4; ++mi)
        __builtin_prefetch(arow[mi] + kc + KC_, 0, 3);
    }
#pragma unroll
    for (int ks = 0; ks < KC_; ks += 32) {
      const int k0 = kc + ks;
      v16bf a0 = load_frag(arow[0] + k0, g);
      v16bf a1 = load_frag(arow[1] + k0, g);
      v16bf a2 = load_frag(arow[2] + k0, g);
      v16bf a3 = load_frag(arow[3] + k0, g);
#pragma unroll
      for (int j = 0; j < 4; ++j) {
        const __bf16* brow = bbuf + (j * 16 + idx) * SBS_ + ks;
        v16bf b = load_frag(brow, g);
        acc[0][j] = __builtin_amdgcn_wmma_f32_16x16x32_bf16(
            false, a0, false, b, (short)0, acc[0][j], false, false);
        acc[1][j] = __builtin_amdgcn_wmma_f32_16x16x32_bf16(
            false, a1, false, b, (short)0, acc[1][j], false, false);
        acc[2][j] = __builtin_amdgcn_wmma_f32_16x16x32_bf16(
            false, a2, false, b, (short)0, acc[2][j], false, false);
        acc[3][j] = __builtin_amdgcn_wmma_f32_16x16x32_bf16(
            false, a3, false, b, (short)0, acc[3][j], false, false);
      }
    }
    __syncthreads();
  }

  // C layout: VGPR jj -> row (+ jj + 8*g), col (n-tile base + idx)
#pragma unroll
  for (int mi = 0; mi < 4; ++mi) {
#pragma unroll
    for (int j = 0; j < 4; ++j) {
      const int col = n0 + j * 16 + idx;
#pragma unroll
      for (int jj = 0; jj < 8; ++jj) {
        int row = m0 + mi * 16 + jj + 8 * g;
        float val = acc[mi][j][jj];
        if (bias) val += bias[col];
        if (act == 1) val = tanhf(val);
        else if (act == 2) val = __expf(-__expf(val));  // RWKV decay
        size_t off = (size_t)row * N + col;
        if (Cbf) Cbf[off] = (__bf16)val;
        else     Cf[off]  = val;
      }
    }
  }
}

// ---------------------------------------------------------------------------
// Recurrent scan, k-split across blocks. Block = (b,h,slice); lane owns one
// v-column; KSL_=32 floats of state per lane in VGPRs. r/k/decay for the
// slice staged through LDS each step (decay precomputed by the w-GEMM).
//   kv = k[k]*v ; o += r[k]*(h[k] + u[k]*kv) ; h[k] = h[k]*dec[k] + kv
// Partial outputs (per slice) reduced deterministically in post_kernel.
// ---------------------------------------------------------------------------
__global__ __launch_bounds__(256) void scan_kernel(
    const float* __restrict__ rbuf, const float* __restrict__ kbuf,
    const float* __restrict__ decbuf, const float* __restrict__ vbuf,
    const float* __restrict__ bonus, float* __restrict__ opart) {
  const int sl = blockIdx.x % SPLIT_;
  const int bh = blockIdx.x / SPLIT_;
  const int b = bh / H_, h = bh % H_;
  const int kbase = sl * KSL_;
  const int vcol = threadIdx.x;
  __shared__ float sr[KSL_], sk[KSL_], sdec[KSL_], su[KSL_];
  if (threadIdx.x < KSL_)
    su[threadIdx.x] = bonus[h * DK_ + kbase + threadIdx.x];
  float hst[KSL_];
#pragma unroll
  for (int i = 0; i < KSL_; ++i) hst[i] = 0.f;

  float* oslice = opart + ((size_t)(sl * (B_ * H_) + bh)) * T_ * DV_;
  for (int t = 0; t < T_; ++t) {
    size_t rk = ((size_t)(b * T_ + t)) * KD_ + h * DK_ + kbase;
    if (threadIdx.x < KSL_) {
      sr[threadIdx.x] = rbuf[rk + threadIdx.x];
      sk[threadIdx.x] = kbuf[rk + threadIdx.x];
      sdec[threadIdx.x] = decbuf[rk + threadIdx.x];
    }
    __syncthreads();
    float vv = vbuf[((size_t)(b * T_ + t)) * VD_ + h * DV_ + vcol];
    float o = 0.f;
#pragma unroll
    for (int kk = 0; kk < KSL_; ++kk) {
      float kv = sk[kk] * vv;
      o += sr[kk] * (hst[kk] + su[kk] * kv);
      hst[kk] = hst[kk] * sdec[kk] + kv;
    }
    oslice[(size_t)t * DV_ + vcol] = o;
    __syncthreads();
  }
}

// ---------------------------------------------------------------------------
// Post-kernel: sum the SPLIT_ scan partials (fixed order -> deterministic),
// gather heads -> [B,T,VD], LayerNorm, swish-gate with g, emit bf16 rows.
// One block per token.
// ---------------------------------------------------------------------------
__global__ __launch_bounds__(256) void post_kernel(
    const float* __restrict__ opart, const float* __restrict__ gbuf,
    const float* __restrict__ gn_w, const float* __restrict__ gn_b,
    __bf16* __restrict__ gated) {
  const int blk = blockIdx.x;
  const int b = blk / T_, t = blk % T_;
  __shared__ float so[VD_];
  __shared__ float red[256];
  const size_t slice_stride = (size_t)(B_ * H_) * T_ * DV_;
  for (int i = threadIdx.x; i < VD_; i += 256) {
    int h = i / DV_, dv = i % DV_;
    size_t base = ((size_t)(b * H_ + h)) * T_ * DV_ + (size_t)t * DV_ + dv;
    float acc = 0.f;
#pragma unroll
    for (int sl = 0; sl < SPLIT_; ++sl)
      acc += opart[sl * slice_stride + base];
    so[i] = acc;
  }
  __syncthreads();
  float s = 0.f;
  for (int i = threadIdx.x; i < VD_; i += 256) s += so[i];
  red[threadIdx.x] = s; __syncthreads();
  for (int off = 128; off > 0; off >>= 1) {
    if (threadIdx.x < off) red[threadIdx.x] += red[threadIdx.x + off];
    __syncthreads();
  }
  float mean = red[0] / VD_;
  __syncthreads();
  float vs = 0.f;
  for (int i = threadIdx.x; i < VD_; i += 256) {
    float d = so[i] - mean; vs += d * d;
  }
  red[threadIdx.x] = vs; __syncthreads();
  for (int off = 128; off > 0; off >>= 1) {
    if (threadIdx.x < off) red[threadIdx.x] += red[threadIdx.x + off];
    __syncthreads();
  }
  float rstd = rsqrtf(red[0] / VD_ + 1e-5f);
  for (int i = threadIdx.x; i < VD_; i += 256) {
    float gv = gbuf[(size_t)blk * VD_ + i];
    float sw = gv / (1.f + __expf(-gv));            // g * sigmoid(g)
    float val = (so[i] - mean) * rstd * gn_w[i] + gn_b[i];
    gated[(size_t)blk * VD_ + i] = (__bf16)(val * sw);
  }
}

// ---------------------------------------------------------------------------
// Host launcher
// ---------------------------------------------------------------------------
extern "C" void kernel_launch(void* const* d_in, const int* in_sizes, int n_in,
                              void* d_out, int out_size, void* d_ws,
                              size_t ws_size, hipStream_t stream) {
  const float* x      = (const float*)d_in[0];
  const float* xp_mu  = (const float*)d_in[1];
  const float* xp_W1  = (const float*)d_in[2];
  const float* xp_W2  = (const float*)d_in[3];
  const float* x_bias = (const float*)d_in[4];
  const float* r_W    = (const float*)d_in[5];
  const float* w_A    = (const float*)d_in[6];
  const float* w_B    = (const float*)d_in[7];
  const float* w_b    = (const float*)d_in[8];
  const float* k_W    = (const float*)d_in[9];
  const float* v_W    = (const float*)d_in[10];
  const float* g_A    = (const float*)d_in[11];
  const float* g_B    = (const float*)d_in[12];
  const float* g_b    = (const float*)d_in[13];
  const float* bonus  = (const float*)d_in[14];
  const float* gn_w   = (const float*)d_in[15];
  const float* gn_b   = (const float*)d_in[16];
  const float* o_W    = (const float*)d_in[17];

  char* ws = (char*)d_ws;
  size_t off = 0;
  auto alloc = [&](size_t bytes) -> void* {
    off = (off + 255) & ~(size_t)255;
    void* p = ws + off;
    off += bytes;
    return p;
  };
  __bf16* Ar = (__bf16*)alloc((size_t)BT_ * D_ * 2);
  __bf16* Aw = (__bf16*)alloc((size_t)BT_ * D_ * 2);
  __bf16* Ak = (__bf16*)alloc((size_t)BT_ * D_ * 2);
  __bf16* Av = (__bf16*)alloc((size_t)BT_ * D_ * 2);
  __bf16* Ag = (__bf16*)alloc((size_t)BT_ * D_ * 2);
  __bf16* rWb = (__bf16*)alloc((size_t)KD_ * D_ * 2);
  __bf16* kWb = (__bf16*)alloc((size_t)KD_ * D_ * 2);
  __bf16* vWb = (__bf16*)alloc((size_t)VD_ * D_ * 2);
  __bf16* wAb = (__bf16*)alloc((size_t)LR_ * D_ * 2);
  __bf16* gAb = (__bf16*)alloc((size_t)LR_ * D_ * 2);
  __bf16* wBb = (__bf16*)alloc((size_t)KD_ * LR_ * 2);
  __bf16* gBb = (__bf16*)alloc((size_t)VD_ * LR_ * 2);
  __bf16* oWb = (__bf16*)alloc((size_t)D_ * VD_ * 2);
  __bf16* wt    = (__bf16*)alloc((size_t)BT_ * LR_ * 2);
  __bf16* gt    = (__bf16*)alloc((size_t)BT_ * LR_ * 2);
  __bf16* gated = (__bf16*)alloc((size_t)BT_ * VD_ * 2);
  float* rbuf   = (float*)alloc((size_t)BT_ * KD_ * 4);
  float* kbuf   = (float*)alloc((size_t)BT_ * KD_ * 4);
  float* decbuf = (float*)alloc((size_t)BT_ * KD_ * 4);
  float* vbuf   = (float*)alloc((size_t)BT_ * VD_ * 4);
  float* gbuf   = (float*)alloc((size_t)BT_ * VD_ * 4);
  float* opart  = (float*)alloc((size_t)SPLIT_ * B_ * H_ * T_ * DV_ * 4);

  auto cvt = [&](const float* src, __bf16* dst, int n) {
    cvt_f32_bf16<<<(n + 255) / 256, 256, 0, stream>>>(src, dst, n);
  };
  cvt(r_W, rWb, KD_ * D_);
  cvt(k_W, kWb, KD_ * D_);
  cvt(v_W, vWb, VD_ * D_);
  cvt(w_A, wAb, LR_ * D_);
  cvt(g_A, gAb, LR_ * D_);
  cvt(w_B, wBb, KD_ * LR_);
  cvt(g_B, gBb, VD_ * LR_);
  cvt(o_W, oWb, D_ * VD_);

  pre_kernel<<<BT_, 256, 0, stream>>>(x, xp_mu, xp_W1, xp_W2, x_bias,
                                      Ar, Aw, Ak, Av, Ag);

  auto gemm = [&](const __bf16* A, int lda, const __bf16* W, int ldw,
                  float* Cf, __bf16* Cbf, const float* bias,
                  int M, int N, int K, int act) {
    dim3 grid(N / 64, M / 256);
    gemm_bf16_wmma<<<grid, 128, 0, stream>>>(A, lda, W, ldw, Cf, Cbf, bias,
                                             M, N, K, act);
  };
  gemm(Ar, D_, rWb, D_, rbuf, nullptr, nullptr, BT_, KD_, D_, 0);
  gemm(Ak, D_, kWb, D_, kbuf, nullptr, nullptr, BT_, KD_, D_, 0);
  gemm(Av, D_, vWb, D_, vbuf, nullptr, nullptr, BT_, VD_, D_, 0);
  gemm(Aw, D_, wAb, D_, nullptr, wt, nullptr, BT_, LR_, D_, 1);
  gemm(wt, LR_, wBb, LR_, decbuf, nullptr, w_b, BT_, KD_, LR_, 2);  // decay
  gemm(Ag, D_, gAb, D_, nullptr, gt, nullptr, BT_, LR_, D_, 1);
  gemm(gt, LR_, gBb, LR_, gbuf, nullptr, g_b, BT_, VD_, LR_, 0);

  scan_kernel<<<B_ * H_ * SPLIT_, 256, 0, stream>>>(rbuf, kbuf, decbuf, vbuf,
                                                    bonus, opart);
  post_kernel<<<BT_, 256, 0, stream>>>(opart, gbuf, gn_w, gn_b, gated);
  gemm(gated, VD_, oWb, VD_, (float*)d_out, nullptr, nullptr, BT_, D_, VD_, 0);

  (void)in_sizes; (void)n_in; (void)out_size; (void)ws_size;
}
